// GeoTransformer_30099130810414
// MI455X (gfx1250) — matrix operand order repaired and checked by reading, not compiled, not run on hardware
//
#include <hip/hip_runtime.h>
#include <math.h>

// ---------------------------------------------------------------------------
// GeoTransformer (LoFTR-style linear attention) for MI455X / gfx1250.
// All dense projections run on v_wmma_f32_16x16x32_bf16 (fp32 accumulate).
// Shapes fixed by setup_inputs(): B=2, L=64*64=4096, C=256, H=8, D=32,
// K(keypoints)=9, 4 layers (self, cross, self, cross), scale=8, h=w=64.
// ---------------------------------------------------------------------------

typedef __bf16 bf16;
typedef bf16 v8bf  __attribute__((ext_vector_type(8)));
typedef bf16 v16bf __attribute__((ext_vector_type(16)));
typedef float v8f  __attribute__((ext_vector_type(8)));

#define B_       2
#define L_       4096
#define C_       256
#define H_       8
#define D_       32
#define KP_      9
#define ROWS_    (B_ * L_)        /* 8192  */
#define SRCROWS_ (ROWS_ * KP_)    /* 73728 */
#define ATTN_EPS 1e-6f
#define LN_EPS   1e-5f

// ---------------- bf16 helpers (RNE) ----------------
static __device__ __forceinline__ float bf2f(bf16 x) {
  unsigned short s = __builtin_bit_cast(unsigned short, x);
  unsigned int u = ((unsigned int)s) << 16;
  return __builtin_bit_cast(float, u);
}
static __device__ __forceinline__ bf16 f2bf(float f) {
  unsigned int u = __builtin_bit_cast(unsigned int, f);
  u += 0x7FFFu + ((u >> 16) & 1u);
  unsigned short s = (unsigned short)(u >> 16);
  return __builtin_bit_cast(bf16, s);
}
static __device__ __forceinline__ float elu1(float x) {
  // elu(x)+1 ; alpha=1  ->  x>0 ? x+1 : exp(x)
  return x > 0.f ? x + 1.f : expf(x);
}

// ---------------- elementwise kernels ----------------
__global__ void cvt_bf_kernel(const float* __restrict__ in, bf16* __restrict__ out, int n) {
  int i = blockIdx.x * 256 + threadIdx.x;
  if (i < n) out[i] = f2bf(in[i]);
}

// W [K,N] fp32 row-major  ->  Wt [N,K] bf16 row-major
__global__ void wtrans_kernel(const float* __restrict__ W, bf16* __restrict__ Wt, int K, int N) {
  int i = blockIdx.x * 256 + threadIdx.x;
  if (i >= K * N) return;
  int k = i / N, n = i - k * N;
  Wt[(size_t)n * K + k] = f2bf(W[i]);
}

__global__ void tanh_bf_kernel(const float* __restrict__ in, bf16* __restrict__ out, int n) {
  int i = blockIdx.x * 256 + threadIdx.x;
  if (i < n) out[i] = f2bf(tanhf(in[i]));
}

// cat[r, 0:256]=x[r,:], cat[r,256:512]=m[r,:]  (bf16 out)
__global__ void concat_bf_kernel(const float* __restrict__ x, const float* __restrict__ m,
                                 bf16* __restrict__ cat) {
  int i = blockIdx.x * 256 + threadIdx.x;     // over ROWS_*512
  int r = i >> 9, c = i & 511;
  float v = (c < C_) ? x[(size_t)r * C_ + c] : m[(size_t)r * C_ + (c - C_)];
  cat[i] = f2bf(v);
}

// ---------------- WMMA GEMM: C[M,N] = A[M,K](bf16) @ Wt[N,K]^T(bf16) ----------------
// block = 256 threads = 8 waves; wave tile 32x64 (2x4 wmma frags -> 8 wmma per
// K-step, 12 b128 loads); block tile 128x128.
template <bool STORE_BF>
__global__ __launch_bounds__(256) void gemm_bf16_kernel(
    const bf16* __restrict__ A, const bf16* __restrict__ Wt,
    float* __restrict__ Cf, bf16* __restrict__ Cb, int M, int N, int K) {
  const int lane = threadIdx.x & 31;
  const int wid  = threadIdx.x >> 5;
  const int half = lane >> 4;      // 0/1 : lane group
  const int lrow = lane & 15;
  const int mBase = blockIdx.y * 128 + (wid & 3) * 32;
  const int nBase = blockIdx.x * 128 + (wid >> 2) * 64;

  const bf16* ap0 = A + (size_t)(mBase + lrow) * K;
  const bf16* ap1 = A + (size_t)(mBase + 16 + lrow) * K;
  const bf16* bp[4];
#pragma unroll
  for (int j = 0; j < 4; ++j) bp[j] = Wt + (size_t)(nBase + j * 16 + lrow) * K;

  v8f acc[2][4] = {};

  for (int k0 = 0; k0 < K; k0 += 32) {
    if (k0 + 32 < K) {  // emits global_prefetch_b8 for the next K slice
      __builtin_prefetch(ap0 + k0 + 32, 0, 1);
      __builtin_prefetch(ap1 + k0 + 32, 0, 1);
      __builtin_prefetch(bp[0] + k0 + 32, 0, 1);
      __builtin_prefetch(bp[2] + k0 + 32, 0, 1);
    }
    // A 16x32 bf16 fragment: lanes 0-15 rows M, contiguous 16 k-values at half*16
    v16bf a[2];
    {
      a[0] = *(const v16bf*)(ap0 + k0 + half * 16);
      a[1] = *(const v16bf*)(ap1 + k0 + half * 16);
    }
    // B 32x16 fragments from Wt[N,K]: lane = n, contiguous 16 k-values at half*16
    v16bf b[4];
#pragma unroll
    for (int j = 0; j < 4; ++j) b[j] = *(const v16bf*)(bp[j] + k0 + half * 16);

#pragma unroll
    for (int i = 0; i < 2; ++i)
#pragma unroll
      for (int j = 0; j < 4; ++j)
        acc[i][j] = __builtin_amdgcn_wmma_f32_16x16x32_bf16(
            false, a[i], false, b[j], (short)0, acc[i][j], false, false);
  }

  // C/D layout: vgpr r -> m = r + half*8 ; n = lrow
#pragma unroll
  for (int i = 0; i < 2; ++i) {
    const int m0 = mBase + i * 16 + half * 8;
#pragma unroll
    for (int j = 0; j < 4; ++j) {
      const int n0 = nBase + j * 16 + lrow;
#pragma unroll
      for (int r = 0; r < 8; ++r) {
        if constexpr (STORE_BF) {
          Cb[(size_t)(m0 + r) * N + n0] = f2bf(acc[i][j][r]);
        } else {
          Cf[(size_t)(m0 + r) * N + n0] = acc[i][j][r];
        }
      }
    }
  }
}

// ---------------- self-attention: KV[n,h,d,e] & Ksum[n,h,d] reduction ----------------
// one block per (n,h); K' = (elu(k)+1)*mask ; V' = v*mask/L
__global__ void kv_reduce_kernel(const float* __restrict__ k, const float* __restrict__ v,
                                 const unsigned char* __restrict__ mask,
                                 float* __restrict__ KV, float* __restrict__ Ksum) {
  const int n = blockIdx.x >> 3, h = blockIdx.x & 7;
  const int t = threadIdx.x;
  __shared__ float lk[8][D_];
  __shared__ float lv[8][D_];
  const int srow = t >> 5, d = t & 31;
  const int dd = t >> 3, e0 = (t & 7) * 4;   // this thread owns KV[dd][e0..e0+3]
  float a0 = 0.f, a1 = 0.f, a2 = 0.f, a3 = 0.f, ks = 0.f;
  const float invL = 1.0f / (float)L_;

  for (int s0 = 0; s0 < L_; s0 += 8) {
    const int s = s0 + srow;
    const float m = (float)mask[n * L_ + s];
    const size_t base = ((size_t)(n * L_ + s)) * C_ + h * D_ + d;
    lk[srow][d] = elu1(k[base]) * m;
    lv[srow][d] = v[base] * m * invL;
    __syncthreads();
#pragma unroll
    for (int ss = 0; ss < 8; ++ss) {
      const float kk = lk[ss][dd];
      a0 += kk * lv[ss][e0 + 0];
      a1 += kk * lv[ss][e0 + 1];
      a2 += kk * lv[ss][e0 + 2];
      a3 += kk * lv[ss][e0 + 3];
    }
    if (t < D_) {
#pragma unroll
      for (int ss = 0; ss < 8; ++ss) ks += lk[ss][t];
    }
    __syncthreads();
  }
  const int bo = n * H_ + h;
  float* kvp = KV + (size_t)bo * D_ * D_ + dd * D_ + e0;
  kvp[0] = a0; kvp[1] = a1; kvp[2] = a2; kvp[3] = a3;
  if (t < D_) Ksum[bo * D_ + t] = ks;
}

// per-row output: msg[r,h,e] = (sum_d Q[d]*KV[d,e]) / (Q.Ksum + eps) * L  (bf16 out)
__global__ void lin_attn_out_kernel(const float* __restrict__ q, const float* __restrict__ KV,
                                    const float* __restrict__ Ksum, bf16* __restrict__ msg) {
  const int r = blockIdx.x, t = threadIdx.x;
  __shared__ float lq[C_];
  lq[t] = elu1(q[(size_t)r * C_ + t]);
  __syncthreads();
  const int h = t >> 5, e = t & 31;
  const int bo = (r >> 12) * H_ + h;   // n = r / L_
  const float* kvp = KV + (size_t)bo * D_ * D_;
  const float* ksp = Ksum + bo * D_;
  float zdot = 0.f, out = 0.f;
#pragma unroll
  for (int d = 0; d < D_; ++d) {
    const float qd = lq[h * D_ + d];
    zdot += qd * ksp[d];
    out  += qd * kvp[d * D_ + e];
  }
  msg[(size_t)r * C_ + t] = f2bf(out * (1.f / (zdot + ATTN_EPS)) * (float)L_);
}

// ---------------- layernorm (+ optional residual), one block per row ----------------
__global__ void layernorm_kernel(const float* __restrict__ in, const float* __restrict__ g,
                                 const float* __restrict__ b, const float* __restrict__ res,
                                 float* __restrict__ out) {
  const int r = blockIdx.x, t = threadIdx.x;
  __shared__ float red[256];
  const float x = in[(size_t)r * C_ + t];
  red[t] = x;
  __syncthreads();
  for (int s = 128; s > 0; s >>= 1) { if (t < s) red[t] += red[t + s]; __syncthreads(); }
  const float mean = red[0] * (1.f / C_);
  __syncthreads();
  const float dx = x - mean;
  red[t] = dx * dx;
  __syncthreads();
  for (int s = 128; s > 0; s >>= 1) { if (t < s) red[t] += red[t + s]; __syncthreads(); }
  const float var = red[0] * (1.f / C_);
  float y = dx * rsqrtf(var + LN_EPS) * g[t] + b[t];
  if (res) y += res[(size_t)r * C_ + t];
  out[(size_t)r * C_ + t] = y;
}

// ---------------- bilinear descriptor sampling + L2 norm ----------------
// one block per (b,l,kp); thread = channel; feat is [B,L,C] with spatial idx y*64+x
__global__ void sample_desc_kernel(const float* __restrict__ feat, const float* __restrict__ kp,
                                   bf16* __restrict__ out) {
  const int rk = blockIdx.x;          // (b*L + l)*KP + k
  const int t  = threadIdx.x;
  const int b  = rk / (L_ * KP_);
  const float kx = kp[(size_t)rk * 2 + 0];
  const float ky = kp[(size_t)rk * 2 + 1];
  const float S = 8.f, Wd = 64.f, Hd = 64.f;
  const float ax = kx - S * 0.5f + 0.5f;
  const float ay = ky - S * 0.5f + 0.5f;
  const float gx = (ax / (Wd * S - S * 0.5f - 0.5f)) * 2.f - 1.f;
  const float gy = (ay / (Hd * S - S * 0.5f - 0.5f)) * 2.f - 1.f;
  const float px = (gx + 1.f) * 0.5f * (Wd - 1.f);
  const float py = (gy + 1.f) * 0.5f * (Hd - 1.f);
  const float x0 = floorf(px), y0 = floorf(py);
  const float wx = px - x0, wy = py - y0;
  int x0i = (int)x0;       x0i = x0i < 0 ? 0 : (x0i > 63 ? 63 : x0i);
  int x1i = (int)x0 + 1;   x1i = x1i < 0 ? 0 : (x1i > 63 ? 63 : x1i);
  int y0i = (int)y0;       y0i = y0i < 0 ? 0 : (y0i > 63 ? 63 : y0i);
  int y1i = (int)y0 + 1;   y1i = y1i < 0 ? 0 : (y1i > 63 ? 63 : y1i);
  const float* fb = feat + (size_t)b * L_ * C_;
  const float g00 = fb[(size_t)(y0i * 64 + x0i) * C_ + t];
  const float g01 = fb[(size_t)(y0i * 64 + x1i) * C_ + t];
  const float g10 = fb[(size_t)(y1i * 64 + x0i) * C_ + t];
  const float g11 = fb[(size_t)(y1i * 64 + x1i) * C_ + t];
  const float val = (1.f - wy) * ((1.f - wx) * g00 + wx * g01)
                  +        wy  * ((1.f - wx) * g10 + wx * g11);
  __shared__ float red[256];
  red[t] = val * val;
  __syncthreads();
  for (int s = 128; s > 0; s >>= 1) { if (t < s) red[t] += red[t + s]; __syncthreads(); }
  const float nrm = sqrtf(red[0]) + 1e-8f;
  out[(size_t)rk * C_ + t] = f2bf(val / nrm);
}

// ---------------- cross attention (1 query, 9 keys), one block per row ----------------
// out[e] = sum_s w_s * v[s,e] / (sum_s w_s + eps),  w_s = Q'.K'[s]  (the /9*9 cancels)
__global__ void cross_attn_kernel(const float* __restrict__ q, const bf16* __restrict__ kb,
                                  const bf16* __restrict__ vb,
                                  const unsigned char* __restrict__ mask,
                                  bf16* __restrict__ msg) {
  const int r = blockIdx.x, t = threadIdx.x;
  __shared__ float lq[C_];
  __shared__ float lk[KP_][C_];
  __shared__ float lv[KP_][C_];
  lq[t] = elu1(q[(size_t)r * C_ + t]);
  for (int s = 0; s < KP_; ++s) {
    const float m = (float)mask[r * KP_ + s];
    const size_t base = ((size_t)(r * KP_ + s)) * C_ + t;
    lk[s][t] = elu1(bf2f(kb[base])) * m;
    lv[s][t] = bf2f(vb[base]) * m;
  }
  __syncthreads();
  const int h = t >> 5, e = t & 31;
  const float* qh = lq + h * D_;
  float acc = 0.f, wsum = 0.f;
#pragma unroll
  for (int s = 0; s < KP_; ++s) {
    float w = 0.f;
#pragma unroll
    for (int d = 0; d < D_; ++d) w += qh[d] * lk[s][h * D_ + d];
    wsum += w;
    acc  += w * lv[s][h * D_ + e];
  }
  msg[(size_t)r * C_ + t] = f2bf(acc * (1.f / (wsum + ATTN_EPS)));
}

// ---------------------------------------------------------------------------
extern "C" void kernel_launch(void* const* d_in, const int* in_sizes, int n_in,
                              void* d_out, int out_size, void* d_ws, size_t ws_size,
                              hipStream_t stream) {
  (void)in_sizes; (void)n_in; (void)out_size; (void)ws_size;

  const float* feat0 = (const float*)d_in[0];
  const float* feat1 = (const float*)d_in[1];
  const float* kp0   = (const float*)d_in[2];
  const float* kp1   = (const float*)d_in[3];
  const unsigned char* mask_self0  = (const unsigned char*)d_in[4];  // jax bool_ = 1 byte
  const unsigned char* mask_self1  = (const unsigned char*)d_in[5];
  const unsigned char* mask_cross0 = (const unsigned char*)d_in[6];
  const unsigned char* mask_cross1 = (const unsigned char*)d_in[7];
  const float* Wq = (const float*)d_in[8];
  const float* Wk = (const float*)d_in[9];
  const float* Wv = (const float*)d_in[10];
  const float* Wm = (const float*)d_in[11];
  const float* W1 = (const float*)d_in[12];
  const float* W2 = (const float*)d_in[13];
  const float* g1 = (const float*)d_in[14];
  const float* b1 = (const float*)d_in[15];
  const float* g2 = (const float*)d_in[16];
  const float* b2 = (const float*)d_in[17];

  // ---- bump allocator on workspace ----
  char* wsp = (char*)d_ws;
  size_t off = 0;
  auto alloc = [&](size_t bytes) -> void* {
    void* p = wsp + off;
    off += (bytes + 255) & ~(size_t)255;
    return p;
  };
  float* featA = (float*)alloc((size_t)ROWS_ * C_ * 4);
  float* featB = (float*)alloc((size_t)ROWS_ * C_ * 4);
  bf16* WqT = (bf16*)alloc((size_t)C_ * C_ * 2);
  bf16* WkT = (bf16*)alloc((size_t)C_ * C_ * 2);
  bf16* WvT = (bf16*)alloc((size_t)C_ * C_ * 2);
  bf16* WmT = (bf16*)alloc((size_t)C_ * C_ * 2);
  bf16* W1T = (bf16*)alloc((size_t)512 * 512 * 2);
  bf16* W2T = (bf16*)alloc((size_t)256 * 512 * 2);
  bf16*  xbf   = (bf16*)alloc((size_t)ROWS_ * C_ * 2);
  float* qbuf  = (float*)alloc((size_t)ROWS_ * C_ * 4);
  float* kbuf  = (float*)alloc((size_t)ROWS_ * C_ * 4);
  float* vbuf  = (float*)alloc((size_t)ROWS_ * C_ * 4);
  bf16*  srcA  = (bf16*)alloc((size_t)SRCROWS_ * C_ * 2);   // f0c (from feat0's map)
  bf16*  srcB  = (bf16*)alloc((size_t)SRCROWS_ * C_ * 2);   // f1c (from feat1's map)
  bf16*  kbf   = (bf16*)alloc((size_t)SRCROWS_ * C_ * 2);
  bf16*  vbf   = (bf16*)alloc((size_t)SRCROWS_ * C_ * 2);
  float* KVb   = (float*)alloc((size_t)B_ * H_ * D_ * D_ * 4);
  float* Ksumb = (float*)alloc((size_t)B_ * H_ * D_ * 4);
  bf16*  msgbf = (bf16*)alloc((size_t)ROWS_ * C_ * 2);
  float* mtmp  = (float*)alloc((size_t)ROWS_ * C_ * 4);
  float* msg1  = (float*)alloc((size_t)ROWS_ * C_ * 4);
  bf16*  catb  = (bf16*)alloc((size_t)ROWS_ * 512 * 2);
  float* t1    = (float*)alloc((size_t)ROWS_ * 512 * 4);
  bf16*  t1bf  = (bf16*)alloc((size_t)ROWS_ * 512 * 2);
  float* t2    = (float*)alloc((size_t)ROWS_ * C_ * 4);

  hipMemcpyAsync(featA, feat0, (size_t)ROWS_ * C_ * 4, hipMemcpyDeviceToDevice, stream);
  hipMemcpyAsync(featB, feat1, (size_t)ROWS_ * C_ * 4, hipMemcpyDeviceToDevice, stream);

  auto gemm_f32 = [&](const bf16* A, const bf16* Wt, float* Cout, int M, int N, int K) {
    dim3 g((unsigned)(N / 128), (unsigned)(M / 128));
    gemm_bf16_kernel<false><<<g, 256, 0, stream>>>(A, Wt, Cout, nullptr, M, N, K);
  };
  auto gemm_bfout = [&](const bf16* A, const bf16* Wt, bf16* Cout, int M, int N, int K) {
    dim3 g((unsigned)(N / 128), (unsigned)(M / 128));
    gemm_bf16_kernel<true><<<g, 256, 0, stream>>>(A, Wt, nullptr, Cout, M, N, K);
  };
  auto wtrans = [&](const float* W, bf16* Wt, int K, int N) {
    wtrans_kernel<<<(K * N + 255) / 256, 256, 0, stream>>>(W, Wt, K, N);
  };

  // msgbf -> @Wm -> LN(g1,b1) -> FFN(tanh) -> LN(g2,b2) + residual; updates feat in place
  auto tail = [&](float* feat, int layer) {
    gemm_f32(msgbf, WmT, mtmp, ROWS_, C_, C_);
    layernorm_kernel<<<ROWS_, 256, 0, stream>>>(mtmp, g1 + layer * C_, b1 + layer * C_, nullptr, msg1);
    concat_bf_kernel<<<ROWS_ * 512 / 256, 256, 0, stream>>>(feat, msg1, catb);
    gemm_f32(catb, W1T, t1, ROWS_, 512, 512);
    tanh_bf_kernel<<<ROWS_ * 512 / 256, 256, 0, stream>>>(t1, t1bf, ROWS_ * 512);
    gemm_f32(t1bf, W2T, t2, ROWS_, C_, 512);
    layernorm_kernel<<<ROWS_, 256, 0, stream>>>(t2, g2 + layer * C_, b2 + layer * C_, feat, feat);
  };

  auto enc_self = [&](float* feat, const unsigned char* m, int layer) {
    cvt_bf_kernel<<<ROWS_ * C_ / 256, 256, 0, stream>>>(feat, xbf, ROWS_ * C_);
    gemm_f32(xbf, WqT, qbuf, ROWS_, C_, C_);
    gemm_f32(xbf, WkT, kbuf, ROWS_, C_, C_);
    gemm_f32(xbf, WvT, vbuf, ROWS_, C_, C_);
    kv_reduce_kernel<<<B_ * H_, 256, 0, stream>>>(kbuf, vbuf, m, KVb, Ksumb);
    lin_attn_out_kernel<<<ROWS_, 256, 0, stream>>>(qbuf, KVb, Ksumb, msgbf);
    tail(feat, layer);
  };

  auto enc_cross = [&](float* feat, const bf16* src, const unsigned char* m, int layer) {
    cvt_bf_kernel<<<ROWS_ * C_ / 256, 256, 0, stream>>>(feat, xbf, ROWS_ * C_);
    gemm_f32(xbf, WqT, qbuf, ROWS_, C_, C_);
    gemm_bfout(src, WkT, kbf, SRCROWS_, C_, C_);
    gemm_bfout(src, WvT, vbf, SRCROWS_, C_, C_);
    cross_attn_kernel<<<ROWS_, 256, 0, stream>>>(qbuf, kbf, vbf, m, msgbf);
    tail(feat, layer);
  };

  for (int layer = 0; layer < 4; ++layer) {
    wtrans(Wq + (size_t)layer * C_ * C_, WqT, C_, C_);
    wtrans(Wk + (size_t)layer * C_ * C_, WkT, C_, C_);
    wtrans(Wv + (size_t)layer * C_ * C_, WvT, C_, C_);
    wtrans(Wm + (size_t)layer * C_ * C_, WmT, C_, C_);
    wtrans(W1 + (size_t)layer * 512 * 512, W1T, 512, 512);
    wtrans(W2 + (size_t)layer * 512 * 256, W2T, 512, 256);
    if ((layer & 1) == 0) {  // self
      enc_self(featA, mask_self0, layer);
      enc_self(featB, mask_self1, layer);
    } else {                 // cross: sample both maps BEFORE updating either feature
      sample_desc_kernel<<<SRCROWS_, 256, 0, stream>>>(featA, kp0, srcA);  // f0c
      sample_desc_kernel<<<SRCROWS_, 256, 0, stream>>>(featB, kp1, srcB);  // f1c
      enc_cross(featA, srcB, mask_cross1, layer);  // feat0 attends to f1c
      enc_cross(featB, srcA, mask_cross0, layer);  // feat1 attends to f0c
    }
  }

  hipMemcpyAsync(d_out, featA, (size_t)ROWS_ * C_ * 4, hipMemcpyDeviceToDevice, stream);
  hipMemcpyAsync((float*)d_out + (size_t)ROWS_ * C_, featB, (size_t)ROWS_ * C_ * 4,
                 hipMemcpyDeviceToDevice, stream);
}
